// AttentionBlock_90417651515809
// MI455X (gfx1250) — matrix operand order repaired and checked by reading, not compiled, not run on hardware
//
#include <hip/hip_runtime.h>
#include <hip/hip_bf16.h>
#include <stdint.h>

// ---------------------------------------------------------------------------
// Graph attention block for MI455X (gfx1250, wave32, WMMA + TDM).
//   N=50000 nodes, M=800000 edges, H=8 heads, DK=DV=32, DE=256.
// Pipeline:
//   1) convert x / W_qkv / W_ff to bf16
//   2) QKV GEMM via v_wmma_f32_16x16x32_bf16; B panel staged to LDS by the
//      Tensor Data Mover (tensor_load_to_lds, padded rows -> conflict-free
//      ds_load_b128 fragment reads), A fragments double-buffered from global
//   3) edge logits + segment max (int-punned float atomic max)
//   4) exp + segment sum (global_atomic_add_f32)
//   5) weighted scatter of V (wave per edge-head, lane per dim)
//   6) convert aggregate to bf16, output GEMM (same kernel)
// ---------------------------------------------------------------------------

typedef __attribute__((ext_vector_type(16))) __bf16 v16bf;
typedef __attribute__((ext_vector_type(8)))  float  v8f;
typedef __attribute__((ext_vector_type(4)))  unsigned int u32x4;
typedef __attribute__((ext_vector_type(8)))  int i32x8;
typedef __attribute__((ext_vector_type(4)))  int i32x4;

#define HEADS      8
#define DKV        32
#define D_EMB      256
#define QKV_COLS   768     // (2*DK + DV) * H
#define INV_SQRT_DK 0.1767766952966369f

// LDS B panel: 64 rows x 256 K shorts, row stride padded 256 -> 264 shorts
// (TDM pad: +4 DWORDs after every 128 DWORDs). ds_load_b128 banks:
// (ln*132 + c) % 64 over 16 lanes x 4 dwords = all 64 banks, conflict-free.
#define BP_ROWS    64
#define BP_STRIDE  264

// ---------------- helpers ----------------

__device__ __forceinline__ unsigned short f32_to_bf16(float f) {
    unsigned int u = __float_as_uint(f);
    u += 0x7FFFu + ((u >> 16) & 1u);          // round-to-nearest-even
    return (unsigned short)(u >> 16);
}
__device__ __forceinline__ float bf16lo(unsigned int u) { return __uint_as_float(u << 16); }
__device__ __forceinline__ float bf16hi(unsigned int u) { return __uint_as_float(u & 0xFFFF0000u); }
__device__ __forceinline__ float bf16s(unsigned short s) { return __uint_as_float(((unsigned int)s) << 16); }

// sign-magnitude int atomic trick: works for mixed-sign IEEE floats
__device__ __forceinline__ void atomicMaxF32(float* addr, float value) {
    if (value >= 0.0f)
        atomicMax((int*)addr, __float_as_int(value));
    else
        atomicMin((unsigned int*)addr, __float_as_uint(value));
}

// ---------------- elementwise kernels ----------------

__global__ void cvt_f32_to_bf16_kernel(const float* __restrict__ in,
                                       unsigned short* __restrict__ out, int n) {
    int i = blockIdx.x * blockDim.x + threadIdx.x;
    if (i < n) out[i] = f32_to_bf16(in[i]);
}

__global__ void init_kernel(float* __restrict__ agg, float* __restrict__ segmax,
                            float* __restrict__ denom, int nAgg, int nNH) {
    int i = blockIdx.x * blockDim.x + threadIdx.x;
    if (i < nAgg) agg[i] = 0.0f;
    if (i < nNH) { segmax[i] = -3.402823466e38f; denom[i] = 0.0f; }
}

// ---------------- bf16 WMMA GEMM with TDM-staged B panel ----------------
//   C(Mrows x Ncols) = A(Mrows x K) * B(Ncols x K)^T + bias
// Block: 8 waves -> 128 rows x 64 cols. One TDM descriptor stages the
// 64 x K bf16 B panel into LDS (rows padded to BP_STRIDE shorts).
// Fragment layouts per CDNA5 ISA 7.12.2:
//   A 16x32 bf16: lanes 0-15 = rows; v[0..3] = K 0..7 (lo half) / 8..15 (hi),
//                 v[4..7] = +16
//   B 32x16 bf16: lane = column; same K order as A (0..7 lo / 8..15 hi, then +16)
//   C/D f32: vgpr r, lanes 0-15 -> (M=r, N=lane); lanes 16-31 -> (M=r+8)

__global__ void gemm_bf16_wmma_kernel(const unsigned short* __restrict__ A,
                                      const unsigned short* __restrict__ B,
                                      const float* __restrict__ bias,
                                      float* __restrict__ Cf,             // fp32 out (or null)
                                      unsigned short* __restrict__ Cbf,   // bf16 out (or null)
                                      int Mrows, int Ncols, int K) {
    __shared__ unsigned short bs[BP_ROWS * BP_STRIDE];

    const int lane    = threadIdx.x & 31;
    const int wave    = threadIdx.x >> 5;
    const int colGrps = Ncols >> 6;
    const int rg      = blockIdx.x / colGrps;        // row group (128 rows)
    const int colBase = (blockIdx.x % colGrps) << 6; // 64 cols per block

    // ---- stage B panel [colBase .. colBase+63] x [0..K) into LDS via TDM ----
    if (wave == 0) {
        const unsigned int lds_off = (unsigned int)(uintptr_t)(&bs[0]);
        const unsigned long long ga =
            (unsigned long long)(uintptr_t)(B + (size_t)colBase * K);

        u32x4 g0;
        g0[0] = 1u;                                          // count=1 (valid user D#)
        g0[1] = lds_off;                                     // lds_addr (bytes)
        g0[2] = (unsigned int)(ga & 0xFFFFFFFFull);          // global_addr[31:0]
        g0[3] = (unsigned int)((ga >> 32) & 0x1FFFFFFull)    // global_addr[56:32]
              | (2u << 30);                                  // type=2 ("image")

        i32x8 g1;
        g1[0] = (1 << 16)        // data_size = 2 bytes
              | (1 << 20)        // pad_enable
              | (6 << 22)        // pad_interval: 2^(6+1)=128 DWORDs (one 256-short row)
              | (3 << 25);       // pad_amount: 4 DWORDs (8 shorts)
        g1[1] = (int)((K & 0xFFFF) << 16);                   // tensor_dim0[15:0] @ bit48
        g1[2] = (int)(((unsigned)K >> 16) & 0xFFFFu)         // tensor_dim0[31:16]
              | (int)((Ncols & 0xFFFF) << 16);               // tensor_dim1[15:0]
        g1[3] = (int)(((unsigned)Ncols >> 16) & 0xFFFFu)     // tensor_dim1[31:16]
              | (int)((K & 0xFFFF) << 16);                   // tile_dim0 = K
        g1[4] = BP_ROWS;                                     // tile_dim1 = 64, tile_dim2 = 0
        g1[5] = (int)K;                                      // tensor_dim0_stride[31:0]
        g1[6] = 0;                                           // stride0[47:32] | stride1[15:0]
        g1[7] = 0;                                           // stride1[47:16]

        i32x4 g2 = {0, 0, 0, 0};
        i32x4 g3 = {0, 0, 0, 0};
        i32x8 g4 = {0, 0, 0, 0, 0, 0, 0, 0};
        __builtin_amdgcn_tensor_load_to_lds(g0, g1, g2, g3, g4, 0);
        __builtin_amdgcn_s_wait_tensorcnt(0);
    }
    __syncthreads();

    const int rowTile = rg * 8 + wave;
    if (rowTile * 16 >= Mrows) return;                 // Mrows is a multiple of 16 here

    const int half = lane >> 4;                        // 0: lanes 0-15, 1: lanes 16-31
    const int ln   = lane & 15;

    union Frag { uint4 q[2]; v16bf v; };

    v8f zero = {0.f, 0.f, 0.f, 0.f, 0.f, 0.f, 0.f, 0.f};
    v8f acc[4];
    acc[0] = zero; acc[1] = zero; acc[2] = zero; acc[3] = zero;

    const unsigned short* arow = A + (size_t)(rowTile * 16 + ln) * K;

    // software-pipelined A fragments (global), B fragments from LDS
    Frag a_cur;
    a_cur.q[0] = *(const uint4*)(arow + half * 8);
    a_cur.q[1] = *(const uint4*)(arow + 16 + half * 8);

    for (int k0 = 0; k0 < K; k0 += 32) {
        const int kn = (k0 + 32 < K) ? (k0 + 32) : k0;
        Frag a_nxt;
        a_nxt.q[0] = *(const uint4*)(arow + kn + half * 8);
        a_nxt.q[1] = *(const uint4*)(arow + kn + 16 + half * 8);
#pragma unroll
        for (int t = 0; t < 4; ++t) {
            const unsigned short* bsrc = &bs[(t * 16 + ln) * BP_STRIDE + k0 + half * 8];
            Frag b;
            b.q[0] = *(const uint4*)(bsrc);
            b.q[1] = *(const uint4*)(bsrc + 16);
            acc[t] = __builtin_amdgcn_wmma_f32_16x16x32_bf16(
                         false, a_cur.v, false, b.v, (short)0, acc[t], false, false);
        }
        a_cur = a_nxt;
    }

#pragma unroll
    for (int t = 0; t < 4; ++t) {
        const int col = colBase + t * 16 + ln;
        const float bv = bias[col];
#pragma unroll
        for (int r = 0; r < 8; ++r) {
            const int m = rowTile * 16 + half * 8 + r;
            const float v = acc[t][r] + bv;
            if (Cf)  Cf[(size_t)m * Ncols + col] = v;
            else     Cbf[(size_t)m * Ncols + col] = f32_to_bf16(v);
        }
    }
}

// ---------------- edge phase ----------------

// att[e,h] = (Q[r,h,:] . K[s,h,:]) / sqrt(DK);  segmax[r,h] = max(...)
__global__ void edge_logits_kernel(const unsigned short* __restrict__ qkv,
                                   const int* __restrict__ senders,
                                   const int* __restrict__ receivers,
                                   float* __restrict__ att,
                                   float* __restrict__ segmax, int MH) {
    int t = blockIdx.x * blockDim.x + threadIdx.x;
    if (t >= MH) return;
    const int e = t >> 3, h = t & 7;
    const int r = receivers[e], s = senders[e];
    const uint4* qp = (const uint4*)(qkv + (size_t)r * QKV_COLS + h * DKV);
    const uint4* kp = (const uint4*)(qkv + (size_t)s * QKV_COLS + D_EMB + h * DKV);
    float acc = 0.f;
#pragma unroll
    for (int i = 0; i < 4; ++i) {
        uint4 a = qp[i], b = kp[i];
        acc += bf16lo(a.x) * bf16lo(b.x) + bf16hi(a.x) * bf16hi(b.x);
        acc += bf16lo(a.y) * bf16lo(b.y) + bf16hi(a.y) * bf16hi(b.y);
        acc += bf16lo(a.z) * bf16lo(b.z) + bf16hi(a.z) * bf16hi(b.z);
        acc += bf16lo(a.w) * bf16lo(b.w) + bf16hi(a.w) * bf16hi(b.w);
    }
    const float logit = acc * INV_SQRT_DK;
    att[t] = logit;
    atomicMaxF32(&segmax[r * HEADS + h], logit);
}

// att[e,h] <- exp(att - segmax[r,h]);  denom[r,h] += att
__global__ void edge_exp_kernel(const int* __restrict__ receivers,
                                float* __restrict__ att,
                                const float* __restrict__ segmax,
                                float* __restrict__ denom, int MH) {
    int t = blockIdx.x * blockDim.x + threadIdx.x;
    if (t >= MH) return;
    const int e = t >> 3, h = t & 7;
    const int r = receivers[e];
    const float ev = __expf(att[t] - segmax[r * HEADS + h]);
    att[t] = ev;
    atomicAdd(&denom[r * HEADS + h], ev);
}

// one wave per (edge, head); lane = value dim
__global__ void edge_scatter_kernel(const unsigned short* __restrict__ qkv,
                                    const float* __restrict__ eweights,
                                    const float* __restrict__ denom,
                                    const int* __restrict__ senders,
                                    const int* __restrict__ receivers,
                                    float* __restrict__ agg, int MH) {
    const int gw   = blockIdx.x * (blockDim.x >> 5) + (threadIdx.x >> 5);
    const int lane = threadIdx.x & 31;
    if (gw >= MH) return;
    const int e = gw >> 3, h = gw & 7;
    const int r = receivers[e], s = senders[e];
    const float ev = eweights[gw];
    const float dn = denom[r * HEADS + h];
    const float w  = (dn > 0.f) ? (ev / dn) : 0.f;
    const float v  = bf16s(qkv[(size_t)s * QKV_COLS + 2 * D_EMB + h * DKV + lane]);
    atomicAdd(&agg[(size_t)r * D_EMB + h * DKV + lane], w * v);
}

// ---------------- launch ----------------

extern "C" void kernel_launch(void* const* d_in, const int* in_sizes, int n_in,
                              void* d_out, int out_size, void* d_ws, size_t ws_size,
                              hipStream_t stream) {
    const float* x    = (const float*)d_in[0];
    const int*   edge = (const int*)d_in[1];
    const float* Wqkv = (const float*)d_in[2];
    const float* bqkv = (const float*)d_in[3];
    const float* Wff  = (const float*)d_in[4];
    const float* bff  = (const float*)d_in[5];

    const int N = in_sizes[0] / D_EMB;
    const int M = in_sizes[1] / 2;
    const int* senders   = edge;
    const int* receivers = edge + M;
    const int MH = M * HEADS;

    // workspace carve-out (256B aligned)
    char*  ws  = (char*)d_ws;
    size_t off = 0;
    auto carve = [&](size_t bytes) -> void* {
        void* p = ws + off;
        off = (off + bytes + 255) & ~(size_t)255;
        return p;
    };
    unsigned short* xb    = (unsigned short*)carve((size_t)N * D_EMB * 2);
    unsigned short* wqkvb = (unsigned short*)carve((size_t)QKV_COLS * D_EMB * 2);
    unsigned short* wffb  = (unsigned short*)carve((size_t)D_EMB * D_EMB * 2);
    unsigned short* qkvb  = (unsigned short*)carve((size_t)N * QKV_COLS * 2);
    float*          att   = (float*)carve((size_t)MH * 4);
    float*          smax  = (float*)carve((size_t)N * HEADS * 4);
    float*          denom = (float*)carve((size_t)N * HEADS * 4);
    float*          agg   = (float*)carve((size_t)N * D_EMB * 4);
    unsigned short* aggb  = (unsigned short*)carve((size_t)N * D_EMB * 2);

    const int TB = 256;

    // 1) bf16 conversions
    int nX = N * D_EMB;
    cvt_f32_to_bf16_kernel<<<(nX + TB - 1) / TB, TB, 0, stream>>>(x, xb, nX);
    int nWq = QKV_COLS * D_EMB;
    cvt_f32_to_bf16_kernel<<<(nWq + TB - 1) / TB, TB, 0, stream>>>(Wqkv, wqkvb, nWq);
    int nWf = D_EMB * D_EMB;
    cvt_f32_to_bf16_kernel<<<(nWf + TB - 1) / TB, TB, 0, stream>>>(Wff, wffb, nWf);

    // init accumulators
    init_kernel<<<(nX + TB - 1) / TB, TB, 0, stream>>>(agg, smax, denom, nX, N * HEADS);

    // 2) QKV GEMM: (N x 256) @ (768 x 256)^T -> bf16 qkv
    {
        int rowGrps = (N + 127) / 128;
        int blocks  = rowGrps * (QKV_COLS / 64);
        gemm_bf16_wmma_kernel<<<blocks, TB, 0, stream>>>(xb, wqkvb, bqkv,
                                                         (float*)nullptr, qkvb,
                                                         N, QKV_COLS, D_EMB);
    }

    // 3) edge logits + segment max
    edge_logits_kernel<<<(MH + TB - 1) / TB, TB, 0, stream>>>(qkvb, senders, receivers,
                                                              att, smax, MH);
    // 4) exp + segment sum
    edge_exp_kernel<<<(MH + TB - 1) / TB, TB, 0, stream>>>(receivers, att, smax, denom, MH);
    // 5) weighted scatter of V (wave per edge-head)
    {
        int blocks = (MH + (TB / 32) - 1) / (TB / 32);
        edge_scatter_kernel<<<blocks, TB, 0, stream>>>(qkvb, att, denom,
                                                       senders, receivers, agg, MH);
    }

    // 6) output GEMM: (N x 256) @ (256 x 256)^T -> fp32 d_out
    cvt_f32_to_bf16_kernel<<<(nX + TB - 1) / TB, TB, 0, stream>>>(agg, aggb, nX);
    {
        int rowGrps = (N + 127) / 128;
        int blocks  = rowGrps * (D_EMB / 64);
        gemm_bf16_wmma_kernel<<<blocks, TB, 0, stream>>>(aggb, wffb, bff,
                                                         (float*)d_out, (unsigned short*)nullptr,
                                                         N, D_EMB, D_EMB);
    }
}
